// DynamicSparseTransformerBlock_39444979646559
// MI455X (gfx1250) — matrix operand order repaired and checked by reading, not compiled, not run on hardware
//
#include <hip/hip_runtime.h>
#include <hip/hip_bf16.h>

typedef __bf16 bf16_t;
typedef __attribute__((ext_vector_type(16))) __bf16 v16bf;
typedef __attribute__((ext_vector_type(8)))  __bf16 v8bf;
typedef __attribute__((ext_vector_type(4)))  __bf16 v4bf;
typedef __attribute__((ext_vector_type(8)))  float  v8f;

// GCC-style vector (matches the "V4i" builtin parameter type exactly)
typedef int v4i_vs __attribute__((__vector_size__(16)));
typedef __attribute__((address_space(1))) v4i_vs* gptr_v4i;  // global (AS1)
typedef __attribute__((address_space(3))) v4i_vs* sptr_v4i;  // LDS (AS3)

#define WMMA_BF16(a, b, c) \
  __builtin_amdgcn_wmma_f32_16x16x32_bf16(false, (a), false, (b), (short)0, (c), false, false)

#if __has_builtin(__builtin_amdgcn_global_load_async_to_lds_b128)
#define HAVE_ASYNC_LDS 1
#endif

static __device__ inline v16bf cat8(v8bf lo, v8bf hi) {
  return __builtin_shufflevector(lo, hi, 0, 1, 2, 3, 4, 5, 6, 7, 8, 9, 10, 11, 12, 13, 14, 15);
}
static __device__ inline v8f zero8() {
  v8f z;
#pragma unroll
  for (int i = 0; i < 8; ++i) z[i] = 0.0f;
  return z;
}

// ---------------------------------------------------------------------------
// K0: L2-normalize q,k over 192 ch, concat pos (64 ch), cast v; emit bf16
// activations in UNFOLDED row order: m = (b*4096 + block)*4 + pos2x2.
// Loads are lane-coalesced (lane = pixel); stores packed into 16B chunks.
// ---------------------------------------------------------------------------
__global__ void pack_inputs(const float* __restrict__ q, const float* __restrict__ k,
                            const float* __restrict__ v, const float* __restrict__ pos,
                            bf16_t* __restrict__ Aq, bf16_t* __restrict__ Ak,
                            bf16_t* __restrict__ Av) {
  int t = blockIdx.x * blockDim.x + threadIdx.x;  // 0..32767
  int b = t >> 14;
  int pix = t & 16383;
  int y = pix >> 7, x = pix & 127;
  int blk = ((y >> 1) << 6) + (x >> 1);
  int p2 = ((y & 1) << 1) + (x & 1);
  size_t m = ((size_t)b * 4096 + blk) * 4 + p2;

  const float* qb = q + (size_t)b * 192 * 16384 + pix;
  const float* kb = k + (size_t)b * 192 * 16384 + pix;
  float sq = 0.f, sk = 0.f;
  for (int c = 0; c < 192; ++c) {
    float a = qb[(size_t)c * 16384];
    float bb = kb[(size_t)c * 16384];
    sq += a * a;
    sk += bb * bb;
  }
  float iq = 1.0f / (sqrtf(sq) + 1e-10f);
  float ik = 1.0f / (sqrtf(sk) + 1e-10f);

  bf16_t* aqr = Aq + m * 256;
  bf16_t* akr = Ak + m * 256;
  bf16_t* avr = Av + m * 256;
  for (int c = 0; c < 192; c += 8) {
    v8bf aq8, ak8;
#pragma unroll
    for (int u = 0; u < 8; ++u) {
      aq8[u] = (bf16_t)(qb[(size_t)(c + u) * 16384] * iq);
      ak8[u] = (bf16_t)(kb[(size_t)(c + u) * 16384] * ik);
    }
    *(v8bf*)(aqr + c) = aq8;
    *(v8bf*)(akr + c) = ak8;
  }
  for (int j = 0; j < 64; j += 8) {
    v8bf p8;
#pragma unroll
    for (int u = 0; u < 8; ++u) p8[u] = (bf16_t)pos[(size_t)(j + u) * 16384 + pix];
    *(v8bf*)(aqr + 192 + j) = p8;
    *(v8bf*)(akr + 192 + j) = p8;
  }
  const float* vb = v + (size_t)b * 256 * 16384 + pix;
  for (int c = 0; c < 256; c += 8) {
    v8bf v8_;
#pragma unroll
    for (int u = 0; u < 8; ++u) v8_[u] = (bf16_t)vb[(size_t)(c + u) * 16384];
    *(v8bf*)(avr + c) = v8_;
  }
}

// ---------------------------------------------------------------------------
// K1: weights f32 -> bf16, concatenated: Wf|Wg|Wh|Wfp|Wgp
// ---------------------------------------------------------------------------
__global__ void pack_w(const float* __restrict__ Wf, const float* __restrict__ Wg,
                       const float* __restrict__ Wh, const float* __restrict__ Wfp,
                       const float* __restrict__ Wgp, bf16_t* __restrict__ o) {
  int i = blockIdx.x * blockDim.x + threadIdx.x;  // < 212992
  float v;
  if (i < 65536)       v = Wf[i];
  else if (i < 131072) v = Wg[i - 65536];
  else if (i < 196608) v = Wh[i - 131072];
  else if (i < 204800) v = Wfp[i - 196608];
  else                 v = Wgp[i - 204800];
  o[i] = (bf16_t)v;
}

// ---------------------------------------------------------------------------
// K2: projection GEMM  D[M=32768, N] = A[M,256] * W^T[256,N] + bias
//   8 waves / WG, wave = 16 rows x (NACC*16) cols, K stepped by 32.
//   Weights staged into LDS via gfx1250 ASYNC-to-LDS (no VGPR round-trip),
//   completion via s_wait_asynccnt; B-frags read back as ds_load_b128.
//   MODE 0: row-major bf16 out [m][N]   MODE 1: h transposed [blk][ch][pos]
// ---------------------------------------------------------------------------
template <int NACC, int MODE>
__global__ void proj_gemm(const bf16_t* __restrict__ A, const bf16_t* __restrict__ Wb,
                          const float* __restrict__ bias, bf16_t* __restrict__ outb, int N) {
  __shared__ __align__(64) bf16_t sW[64 * 256];
  const int NW = NACC * 16;
  int n0 = blockIdx.y * 64;
  int tid = threadIdx.x;
  const bf16_t* wsrc = Wb + (size_t)n0 * 256;

#ifdef HAVE_ASYNC_LDS
  for (int i = tid * 8; i < NW * 256; i += 256 * 8) {
    __builtin_amdgcn_global_load_async_to_lds_b128(
        (gptr_v4i)(wsrc + i), (sptr_v4i)(sW + i), 0, 0);
  }
#if __has_builtin(__builtin_amdgcn_s_wait_asynccnt)
  __builtin_amdgcn_s_wait_asynccnt(0);
#else
  asm volatile("s_wait_asynccnt 0x0" ::: "memory");
#endif
  __syncthreads();
#else
  for (int i = tid * 8; i < NW * 256; i += 256 * 8)
    *(v8bf*)(sW + i) = *(const v8bf*)(wsrc + i);
  __syncthreads();
#endif

  int wave = tid >> 5, lane = tid & 31;
  int lq = lane & 15, half = lane >> 4;
  int m0 = blockIdx.x * 128 + wave * 16;

  v8f acc[NACC];
#pragma unroll
  for (int nt = 0; nt < NACC; ++nt) acc[nt] = zero8();

  const bf16_t* arow = A + (size_t)(m0 + lq) * 256;

#pragma unroll
  for (int ks = 0; ks < 8; ++ks) {
    int c0 = ks * 32 + half * 16;
    v16bf af = cat8(*(const v8bf*)(arow + c0), *(const v8bf*)(arow + c0 + 8));
#pragma unroll
    for (int nt = 0; nt < NACC; ++nt) {
      const bf16_t* bp = sW + ((nt * 16 + lq) * 256 + ks * 32 + half * 16);
      v16bf bfr = *(const v16bf*)bp;  // 2x ds_load_b128
      acc[nt] = WMMA_BF16(af, bfr, acc[nt]);
    }
  }

#pragma unroll
  for (int nt = 0; nt < NACC; ++nt) {
    int n = n0 + nt * 16 + lq;
    float bs = bias[n];
#pragma unroll
    for (int j = 0; j < 8; ++j) {
      int m = m0 + j + 8 * half;
      float val = acc[nt][j] + bs;
      if (MODE == 0) {
        outb[(size_t)m * N + n] = (bf16_t)val;
      } else {
        outb[((size_t)(m >> 2) * 256 + n) * 4 + (m & 3)] = (bf16_t)val;
      }
    }
  }
}

// ---------------------------------------------------------------------------
// K3: block-sparse attention. One wave per query 2x2 block.
//   scores: 16 WMMAs, mask: 2 WMMAs, value: 16 WMMAs.
//   Query rows replicated 4x into the 16-row A fragment (no garbage rows).
// ---------------------------------------------------------------------------
__global__ void attn_kernel(const bf16_t* __restrict__ f_unf, const bf16_t* __restrict__ g_unf,
                            const bf16_t* __restrict__ fp_unf, const bf16_t* __restrict__ gp_unf,
                            const bf16_t* __restrict__ h_t, const int* __restrict__ idx,
                            float* __restrict__ out, float* __restrict__ oidx,
                            float* __restrict__ conf) {
  __shared__ __align__(64) bf16_t smask[8][16 * 32];
  int tid = threadIdx.x, wave = tid >> 5, lane = tid & 31;
  int lq = lane & 15, half = lane >> 4;
  int nb = blockIdx.x * 8 + wave;  // b*4096 + n
  int b = nb >> 12, n = nb & 4095;
  int h2 = n >> 6, w2 = n & 63;
  const int* idxr = idx + (size_t)nb * 8;

  // gathered key-block for this lane's column in each N-tile
  int kb0 = idxr[lq >> 2];
  int kb1 = idxr[4 + (lq >> 2)];
  int p = lq & 3;
  const bf16_t* g0 = g_unf + ((size_t)(b * 4096 + kb0) * 4 + p) * 256;
  const bf16_t* g1 = g_unf + ((size_t)(b * 4096 + kb1) * 4 + p) * 256;
  const bf16_t* arow = f_unf + ((size_t)nb * 4 + (lq & 3)) * 256;
  // warm the WGP-near cache for the gathered key rows (high locality => WGP
  // scope => prefetch pulls into all cache levels, not just GL2)
  __builtin_prefetch(g0, 0, 3);
  __builtin_prefetch(g0 + 128, 0, 3);
  __builtin_prefetch(g1, 0, 3);
  __builtin_prefetch(g1 + 128, 0, 3);
  __builtin_prefetch(arow, 0, 3);

  // ---- scores = qw . qk^T  (K=256 in 8 steps of 32) ----
  v8f s0 = zero8(), s1 = zero8();
#pragma unroll
  for (int ks = 0; ks < 8; ++ks) {
    int c0 = ks * 32 + half * 16;
    v16bf af = cat8(*(const v8bf*)(arow + c0), *(const v8bf*)(arow + c0 + 8));
    int cb = ks * 32 + half * 16;
    v16bf b0 = *(const v16bf*)(g0 + cb);
    v16bf b1 = *(const v16bf*)(g1 + cb);
    s0 = WMMA_BF16(af, b0, s0);
    s1 = WMMA_BF16(af, b1, s1);
  }

  // ---- mask logits = qpw . qkp^T  (dp = 32 = one K step) ----
  const bf16_t* aprow = fp_unf + ((size_t)nb * 4 + (lq & 3)) * 32;
  v16bf apf = cat8(*(const v8bf*)(aprow + half * 16), *(const v8bf*)(aprow + 8 + half * 16));
  const bf16_t* gp0 = gp_unf + ((size_t)(b * 4096 + kb0) * 4 + p) * 32 + half * 16;
  const bf16_t* gp1 = gp_unf + ((size_t)(b * 4096 + kb1) * 4 + p) * 32 + half * 16;
  v8f m0a = WMMA_BF16(apf, *(const v16bf*)gp0, zero8());
  v8f m1a = WMMA_BF16(apf, *(const v16bf*)gp1, zero8());

  // ---- softmax over 32 keys per query row (rows 0..3 live in VGPR j=0..3) ----
  const float scale = 1.0f / 16.0f;  // 1/sqrt(256)
  float attn0[4], attn1[4], msk0[4], msk1[4];
#pragma unroll
  for (int j = 0; j < 4; ++j) {
    float a0 = s0[j] * scale, a1 = s1[j] * scale;
    float mx = fmaxf(a0, a1);
#pragma unroll
    for (int d = 1; d < 16; d <<= 1) mx = fmaxf(mx, __shfl_xor(mx, d, 32));
    float e0 = __expf(a0 - mx), e1 = __expf(a1 - mx);
    float sm = e0 + e1;
#pragma unroll
    for (int d = 1; d < 16; d <<= 1) sm += __shfl_xor(sm, d, 32);
    float r = 1.0f / sm;
    attn0[j] = e0 * r;
    attn1[j] = e1 * r;
    msk0[j] = (m0a[j] > 0.0f) ? attn0[j] : 0.0f;
    msk1[j] = (m1a[j] > 0.0f) ? attn1[j] : 0.0f;
    float cf = msk0[j] + msk1[j];
#pragma unroll
    for (int d = 1; d < 16; d <<= 1) cf += __shfl_xor(cf, d, 32);
    if (lane == 0) {
      int y = 2 * h2 + (j >> 1), x = 2 * w2 + (j & 1);
      conf[(size_t)b * 16384 + y * 128 + x] = cf;
    }
  }

  // ---- top-8 of attn per row -> full-res key indices ----
#pragma unroll
  for (int r = 0; r < 4; ++r) {
    float v0 = attn0[r], v1 = attn1[r];
    int y = 2 * h2 + (r >> 1), x = 2 * w2 + (r & 1);
    float* orow = oidx + ((size_t)b * 16384 + y * 128 + x) * 8;
    for (int t = 0; t < 8; ++t) {
      float bv;
      int bk;
      if (v0 >= v1) { bv = v0; bk = lq; } else { bv = v1; bk = 16 + lq; }
#pragma unroll
      for (int d = 1; d < 16; d <<= 1) {
        float ov = __shfl_xor(bv, d, 32);
        int ok = __shfl_xor(bk, d, 32);
        if (ov > bv || (ov == bv && ok < bk)) { bv = ov; bk = ok; }
      }
      if (bk == lq) v0 = -3.0e38f;
      else if (bk == (16 + lq)) v1 = -3.0e38f;
      if (lane == 0) {
        int kbw = idxr[bk >> 2];
        int base = 2 * (kbw >> 6) * 128 + 2 * (kbw & 63);
        int off = ((bk & 2) ? 128 : 0) + (bk & 1);
        orow[t] = (float)(base + off);
      }
    }
  }

  // ---- re-layout masked probs C->A fragment via LDS ----
  bf16_t* smw = smask[wave];
#pragma unroll
  for (int i = lane; i < 512; i += 32) smw[i] = (bf16_t)0.0f;
  if (half == 0) {
#pragma unroll
    for (int j = 0; j < 4; ++j) {
      smw[j * 32 + lq] = (bf16_t)msk0[j];
      smw[j * 32 + 16 + lq] = (bf16_t)msk1[j];
    }
  }
  asm volatile("s_wait_dscnt 0x0" ::: "memory");
  v16bf amf = cat8(*(const v8bf*)(smw + lq * 32 + half * 16),
                   *(const v8bf*)(smw + lq * 32 + 8 + half * 16));

  // ---- out = masked . qv  (K=32 keys, N=256 ch in 16 tiles) ----
  int kbs[4];
#pragma unroll
  for (int g = 0; g < 4; ++g) kbs[g] = idxr[half * 4 + g];
  for (int nt = 0; nt < 16; ++nt) {
    int ch = nt * 16 + lq;
    v16bf bfr;
#pragma unroll
    for (int g = 0; g < 4; ++g) {
      v4bf seg = *(const v4bf*)(h_t + ((size_t)(b * 4096 + kbs[g]) * 256 + ch) * 4);
      bfr[g * 4 + 0] = seg[0];
      bfr[g * 4 + 1] = seg[1];
      bfr[g * 4 + 2] = seg[2];
      bfr[g * 4 + 3] = seg[3];
    }
    v8f o = WMMA_BF16(amf, bfr, zero8());
    if (half == 0) {
#pragma unroll
      for (int j = 0; j < 4; ++j) {
        int y = 2 * h2 + (j >> 1), x = 2 * w2 + (j & 1);
        out[((size_t)b * 256 + ch) * 16384 + y * 128 + x] = o[j];
      }
    }
  }
}

// ---------------------------------------------------------------------------
extern "C" void kernel_launch(void* const* d_in, const int* in_sizes, int n_in,
                              void* d_out, int out_size, void* d_ws, size_t ws_size,
                              hipStream_t stream) {
  const float* q   = (const float*)d_in[0];
  const float* k   = (const float*)d_in[1];
  const float* v   = (const float*)d_in[2];
  const float* pos = (const float*)d_in[3];
  /* d_in[4] = seg_map (unused by reference) */
  const int* pidx  = (const int*)d_in[5];
  const float* Wf  = (const float*)d_in[6];
  const float* bf_ = (const float*)d_in[7];
  const float* Wg  = (const float*)d_in[8];
  const float* bg  = (const float*)d_in[9];
  const float* Wh  = (const float*)d_in[10];
  const float* bh  = (const float*)d_in[11];
  const float* Wfp = (const float*)d_in[12];
  const float* bfp = (const float*)d_in[13];
  const float* Wgp = (const float*)d_in[14];
  const float* bgp = (const float*)d_in[15];

  char* ws = (char*)d_ws;
  bf16_t* Aq     = (bf16_t*)(ws + (size_t)0);
  bf16_t* Ak     = (bf16_t*)(ws + (size_t)16777216);
  bf16_t* Av     = (bf16_t*)(ws + (size_t)33554432);
  bf16_t* f_unf  = (bf16_t*)(ws + (size_t)50331648);
  bf16_t* g_unf  = (bf16_t*)(ws + (size_t)67108864);
  bf16_t* h_t    = (bf16_t*)(ws + (size_t)83886080);
  bf16_t* fp_unf = (bf16_t*)(ws + (size_t)100663296);
  bf16_t* gp_unf = (bf16_t*)(ws + (size_t)102760448);
  bf16_t* Wpk    = (bf16_t*)(ws + (size_t)104857600);

  float* out  = (float*)d_out;             // [2,256,128,128]
  float* oidx = out + 8388608;             // [2,16384,8] (values as float)
  float* conf = oidx + 262144;             // [2,1,128,128]

  pack_inputs<<<128, 256, 0, stream>>>(q, k, v, pos, Aq, Ak, Av);
  pack_w<<<832, 256, 0, stream>>>(Wf, Wg, Wh, Wfp, Wgp, Wpk);

  dim3 g4(256, 4), g1(256, 1);
  proj_gemm<4, 0><<<g4, 256, 0, stream>>>(Aq, Wpk + 0,      bf_, f_unf, 256);
  proj_gemm<4, 0><<<g4, 256, 0, stream>>>(Ak, Wpk + 65536,  bg,  g_unf, 256);
  proj_gemm<4, 1><<<g4, 256, 0, stream>>>(Av, Wpk + 131072, bh,  h_t,   256);
  proj_gemm<2, 0><<<g1, 256, 0, stream>>>(Aq, Wpk + 196608, bfp, fp_unf, 32);
  proj_gemm<2, 0><<<g1, 256, 0, stream>>>(Ak, Wpk + 204800, bgp, gp_unf, 32);

  attn_kernel<<<1024, 256, 0, stream>>>(f_unf, g_unf, fp_unf, gp_unf, h_t, pidx,
                                        out, oidx, conf);
}
